// UPNFlow_31817117729086
// MI455X (gfx1250) — matrix operand order repaired and checked by reading, not compiled, not run on hardware
//
#include <hip/hip_runtime.h>
#include <hip/hip_fp16.h>
#include <hip/hip_bf16.h>

#define BATCH 1024
#define DIMX  128
#define HIDN  1024

typedef _Float16 f16;
typedef __attribute__((ext_vector_type(16))) _Float16 v16h;
typedef __attribute__((ext_vector_type(8)))  _Float16 v8h;
typedef __attribute__((ext_vector_type(8)))  float    v8f;

#define LDPAD 40   // 32 K-values + 8 halves pad (16B-aligned rows, staggered banks)

// ---------------------------------------------------------------------------
// CDNA5 async copy: GLOBAL_LOAD_ASYNC_TO_LDS_B128 (ASYNCcnt-tracked DMA of
// 16B/lane into per-lane LDS addresses; no VGPR return). Generic->LDS offset:
// for the LDS aperture the low 32 address bits ARE the LDS byte offset.
// ---------------------------------------------------------------------------
__device__ __forceinline__ void async_copy_b128(f16* lds_dst, const f16* gsrc) {
  unsigned loff = (unsigned)(uintptr_t)lds_dst;
  unsigned long long g = (unsigned long long)(uintptr_t)gsrc;
  asm volatile("global_load_async_to_lds_b128 %0, %1, off"
               :: "v"(loff), "v"(g) : "memory");
}
__device__ __forceinline__ void wait_async0() {
  asm volatile("s_wait_asynccnt 0x0" ::: "memory");
}

// ---------------------------------------------------------------------------
// Fragment loaders per CDNA5 ISA 7.12.2 (wave32).
// A (16x32 f16): lane&15 = M row; lane>>4 selects K groups {hl*8..} and
//   {16+hl*8..} -> two contiguous ds_load_b128.
// B (32x16 f16): lane&15 = N col; lane>>4 selects K groups {hl*8..} and
//   {16+hl*8..} (same as A). B staged transposed [n][k] -> two ds_load_b128.
// ---------------------------------------------------------------------------
__device__ __forceinline__ v16h frag_a_lds(const f16* s, int row0, int lane) {
  const int r = lane & 15, hl = lane >> 4;
  const f16* p = s + (row0 + r) * LDPAD + hl * 8;
  v8h lo = *(const v8h*)(p);
  v8h hi = *(const v8h*)(p + 16);
  return __builtin_shufflevector(lo, hi, 0,1,2,3,4,5,6,7,8,9,10,11,12,13,14,15);
}

__device__ __forceinline__ v16h frag_b_lds(const f16* s, int col0, int lane) {
  const int c = lane & 15, hl = lane >> 4;
  const f16* p = s + (col0 + c) * LDPAD + hl * 8;
  v8h lo = *(const v8h*)(p);
  v8h hi = *(const v8h*)(p + 16);
  return __builtin_shufflevector(lo, hi, 0,1,2,3,4,5,6,7,8,9,10,11,12,13,14,15);
}

// ---------------------------------------------------------------------------
// prep: f32->f16 weight casts, bias fold of the scalar-t input row, mu extract.
// ---------------------------------------------------------------------------
__global__ void prep_kernel(const float* __restrict__ states, const float* __restrict__ tptr,
    const float* __restrict__ dW1, const float* __restrict__ db1,
    const float* __restrict__ dW2, const float* __restrict__ dW3,
    const float* __restrict__ nW1, const float* __restrict__ nb1,
    const float* __restrict__ nW2, const float* __restrict__ nW3,
    f16* __restrict__ muH,
    f16* __restrict__ W1dh, f16* __restrict__ W2dh, f16* __restrict__ W3dh,
    f16* __restrict__ W1nh, f16* __restrict__ W2nh, f16* __restrict__ W3nh,
    float* __restrict__ b1d, float* __restrict__ b1n) {
  const int tid = blockIdx.x * blockDim.x + threadIdx.x;
  const int stride = gridDim.x * blockDim.x;
  const float t = tptr[0];
  for (int i = tid; i < HIDN * HIDN; i += stride) {
    W2dh[i] = (f16)dW2[i];
    W2nh[i] = (f16)nW2[i];
  }
  for (int i = tid; i < DIMX * HIDN; i += stride) {
    W1dh[i] = (f16)dW1[i];
    W1nh[i] = (f16)nW1[i];
    W3dh[i] = (f16)dW3[i];
    W3nh[i] = (f16)nW3[i];
  }
  for (int i = tid; i < HIDN; i += stride) {
    b1d[i] = db1[i] + t * dW1[DIMX * HIDN + i];
    b1n[i] = nb1[i] + t * nW1[DIMX * HIDN + i];
  }
  for (int i = tid; i < BATCH * DIMX; i += stride) {
    muH[i] = (f16)states[(i >> 7) * (2 * DIMX) + (i & 127)];
  }
}

// ---------------------------------------------------------------------------
// Generic pipelined WMMA GEMM: C = act(A @ W + bias)
//   Double-buffered LDS, one barrier per K-step. A-panel: async DMA to LDS.
//   W-panel: register-prefetched, committed transposed to LDS after compute.
//   act 0: outF = v (stride ldo); 1: outH=tanh f16 + outT=1-h^2 f16 (stride N);
//   act 2: outF = softplus(v) (stride ldo).
// ---------------------------------------------------------------------------
__global__ __launch_bounds__(256)
void wmma_gemm_kernel(const f16* __restrict__ A, const f16* __restrict__ W,
                      const float* __restrict__ bias,
                      f16* __restrict__ outH, f16* __restrict__ outT,
                      float* __restrict__ outF,
                      int N, int K, int ldo, int act) {
  __shared__ __align__(16) f16 As[2][128 * LDPAD];
  __shared__ __align__(16) f16 Ws[2][128 * LDPAD];
  const int lane = threadIdx.x & 31;
  const int wave = threadIdx.x >> 5;
  const int m0 = blockIdx.y * 128;
  const int n0 = blockIdx.x * 128;
  const int ar = threadIdx.x >> 1, ac = (threadIdx.x & 1) * 16;  // A-stage coords
  const int wr = threadIdx.x >> 3, wc = (threadIdx.x & 7) * 16;  // W-stage coords

  v8f acc[8] = {};

  // ---- stage panel 0 into buffer 0 ----
  async_copy_b128(&As[0][ar * LDPAD + ac],     A + (size_t)(m0 + ar) * K + ac);
  async_copy_b128(&As[0][ar * LDPAD + ac + 8], A + (size_t)(m0 + ar) * K + ac + 8);
  {
    v8h w0 = *(const v8h*)(W + (size_t)wr * N + n0 + wc);
    v8h w1 = *(const v8h*)(W + (size_t)wr * N + n0 + wc + 8);
#pragma unroll
    for (int j = 0; j < 8; ++j) {
      Ws[0][(wc + j) * LDPAD + wr]     = w0[j];
      Ws[0][(wc + 8 + j) * LDPAD + wr] = w1[j];
    }
  }

  const int nsteps = K >> 5;
  for (int i = 0; i < nsteps; ++i) {
    const int b = i & 1;
    wait_async0();       // this wave's async fills of panel i are in LDS
    __syncthreads();     // everyone's stage(i) complete (dscnt auto-waited)

    v8h nw0 = {}, nw1 = {};
    const bool have_next = (i + 1 < nsteps);
    if (have_next) {
      const int kb = (i + 1) * 32;
      async_copy_b128(&As[b ^ 1][ar * LDPAD + ac],     A + (size_t)(m0 + ar) * K + kb + ac);
      async_copy_b128(&As[b ^ 1][ar * LDPAD + ac + 8], A + (size_t)(m0 + ar) * K + kb + ac + 8);
      nw0 = *(const v8h*)(W + (size_t)(kb + wr) * N + n0 + wc);
      nw1 = *(const v8h*)(W + (size_t)(kb + wr) * N + n0 + wc + 8);
    }

    v16h af = frag_a_lds(As[b], wave * 16, lane);
#pragma unroll
    for (int nt = 0; nt < 8; ++nt) {
      v16h bf = frag_b_lds(Ws[b], nt * 16, lane);
      acc[nt] = __builtin_amdgcn_wmma_f32_16x16x32_f16(
          false, af, false, bf, (short)0, acc[nt], false, false);
    }

    if (have_next) {  // commit next W panel after compute (hides global latency)
#pragma unroll
      for (int j = 0; j < 8; ++j) {
        Ws[b ^ 1][(wc + j) * LDPAD + wr]     = nw0[j];
        Ws[b ^ 1][(wc + 8 + j) * LDPAD + wr] = nw1[j];
      }
    }
  }

  // Epilogue. D layout: VGPR v -> M = v/v+8 (lane half); N = lane&15.
  const int c = lane & 15, hl = lane >> 4;
#pragma unroll
  for (int nt = 0; nt < 8; ++nt) {
    const int n = n0 + nt * 16 + c;
    const float b = bias[n];
#pragma unroll
    for (int v = 0; v < 8; ++v) {
      const int m = m0 + wave * 16 + hl * 8 + v;
      const float val = acc[nt][v] + b;
      if (act == 1) {
        const float h = tanhf(val);
        outH[(size_t)m * N + n] = (f16)h;
        if (outT) outT[(size_t)m * N + n] = (f16)(1.f - h * h);
      } else if (act == 2) {
        outF[(size_t)m * ldo + n] = (val > 20.f) ? val : log1pf(expf(val));
      } else {
        outF[(size_t)m * ldo + n] = val;
      }
    }
  }
}

// ---------------------------------------------------------------------------
// Jacobian-diagonal kernel: one workgroup per sample s.
//   P(s) = W1[:128,:] @ (t1(s)-row-scaled W2)       (128 x 1024)
//   J_diag[s,i] = sum_kk P[i,kk] * t2[s,kk] * W3[kk,i]
// A-panel (W1) is a pure copy -> async DMA; the per-sample t1 scale rides on
// the W2 transpose-stage. 8 N-blocks, fused projection, then sigma_drift and
// log_density_drift.
// ---------------------------------------------------------------------------
__global__ __launch_bounds__(256)
void jac_kernel(const f16* __restrict__ W1h, const f16* __restrict__ W2h,
                const f16* __restrict__ W3h,
                const f16* __restrict__ t1d, const f16* __restrict__ t2d,
                const float* __restrict__ Qdiag, const float* __restrict__ states,
                float* __restrict__ out) {
  __shared__ __align__(16) f16 As[2][128 * LDPAD];
  __shared__ __align__(16) f16 Ws[2][128 * LDPAD];
  __shared__ __align__(16) f16 t1s[HIDN];
  __shared__ __align__(16) f16 t2s[HIDN];
  __shared__ float Jred[128][17];
  __shared__ float ldred[128];

  const int s = blockIdx.x;
  const int lane = threadIdx.x & 31;
  const int wave = threadIdx.x >> 5;
  const int c = lane & 15, hl = lane >> 4;
  const int ar = threadIdx.x >> 1, ac = (threadIdx.x & 1) * 16;
  const int wr = threadIdx.x >> 3, wc = (threadIdx.x & 7) * 16;

  {
    const int i = threadIdx.x * 8;
    if (i < HIDN) {
      *(v8h*)(t1s + i) = *(const v8h*)(t1d + (size_t)s * HIDN + i);
      *(v8h*)(t2s + i) = *(const v8h*)(t2d + (size_t)s * HIDN + i);
    }
  }

  float jacc[8] = {0.f, 0.f, 0.f, 0.f, 0.f, 0.f, 0.f, 0.f};

  for (int jb = 0; jb < 8; ++jb) {       // N-block of 128 kk columns
    v8f acc[8] = {};
    __syncthreads();  // previous jb's compute done (jb=0: t1s/t2s fill visible)

    // ---- stage panel 0 into buffer 0 ----
    async_copy_b128(&As[0][ar * LDPAD + ac],     W1h + (size_t)ar * HIDN + ac);
    async_copy_b128(&As[0][ar * LDPAD + ac + 8], W1h + (size_t)ar * HIDN + ac + 8);
    {
      v8h w0 = *(const v8h*)(W2h + (size_t)wr * HIDN + jb * 128 + wc);
      v8h w1 = *(const v8h*)(W2h + (size_t)wr * HIDN + jb * 128 + wc + 8);
      const f16 ts = t1s[wr];
      v8h s0 = w0 * ts, s1 = w1 * ts;
#pragma unroll
      for (int j = 0; j < 8; ++j) {
        Ws[0][(wc + j) * LDPAD + wr]     = s0[j];
        Ws[0][(wc + 8 + j) * LDPAD + wr] = s1[j];
      }
    }

    for (int i = 0; i < (HIDN >> 5); ++i) {
      const int b = i & 1;
      wait_async0();
      __syncthreads();

      v8h nw0 = {}, nw1 = {};
      f16 nts = (f16)0.f;
      const bool have_next = (i + 1 < (HIDN >> 5));
      if (have_next) {
        const int kb = (i + 1) * 32;
        async_copy_b128(&As[b ^ 1][ar * LDPAD + ac],     W1h + (size_t)ar * HIDN + kb + ac);
        async_copy_b128(&As[b ^ 1][ar * LDPAD + ac + 8], W1h + (size_t)ar * HIDN + kb + ac + 8);
        nw0 = *(const v8h*)(W2h + (size_t)(kb + wr) * HIDN + jb * 128 + wc);
        nw1 = *(const v8h*)(W2h + (size_t)(kb + wr) * HIDN + jb * 128 + wc + 8);
        nts = t1s[kb + wr];
      }

      v16h af = frag_a_lds(As[b], wave * 16, lane);
#pragma unroll
      for (int nt = 0; nt < 8; ++nt) {
        v16h bf = frag_b_lds(Ws[b], nt * 16, lane);
        acc[nt] = __builtin_amdgcn_wmma_f32_16x16x32_f16(
            false, af, false, bf, (short)0, acc[nt], false, false);
      }

      if (have_next) {
        v8h s0 = nw0 * nts, s1 = nw1 * nts;
#pragma unroll
        for (int j = 0; j < 8; ++j) {
          Ws[b ^ 1][(wc + j) * LDPAD + wr]     = s0[j];
          Ws[b ^ 1][(wc + 8 + j) * LDPAD + wr] = s1[j];
        }
      }
    }

    // Fused projection: lane holds P[i,kk], i = 16*wave + hl*8 + v,
    // kk = jb*128 + nt*16 + (lane&15); weight by t2[kk]*W3[kk,i].
#pragma unroll
    for (int nt = 0; nt < 8; ++nt) {
      const int kk = jb * 128 + nt * 16 + c;
      const float t2w = (float)t2s[kk];
#pragma unroll
      for (int v = 0; v < 8; ++v) {
        const int i = wave * 16 + hl * 8 + v;
        jacc[v] += acc[nt][v] * t2w * (float)W3h[(size_t)kk * DIMX + i];
      }
    }
  }

  // Cross-lane reduce: 16 lanes per row i.
#pragma unroll
  for (int v = 0; v < 8; ++v)
    Jred[wave * 16 + hl * 8 + v][c] = jacc[v];
  __syncthreads();
  if (threadIdx.x < 128) {
    const int i = threadIdx.x;
    float J = 0.f;
#pragma unroll
    for (int j = 0; j < 16; ++j) J += Jred[i][j];
    const float sig = states[(size_t)s * (2 * DIMX) + DIMX + i];
    const float Qd  = Qdiag[(size_t)s * DIMX + i];
    out[(size_t)s * (2 * DIMX) + DIMX + i] = 2.f * J * sig + Qd;   // sigma_drift
    ldred[i] = -J + 0.5f * Qd / (sig + 1e-6f);
  }
  __syncthreads();
  for (int off = 64; off > 0; off >>= 1) {
    if (threadIdx.x < off) ldred[threadIdx.x] += ldred[threadIdx.x + off];
    __syncthreads();
  }
  if (threadIdx.x == 0) out[(size_t)BATCH * (2 * DIMX) + s] = ldred[0];
}

// ---------------------------------------------------------------------------
extern "C" void kernel_launch(void* const* d_in, const int* in_sizes, int n_in,
                              void* d_out, int out_size, void* d_ws, size_t ws_size,
                              hipStream_t stream) {
  (void)in_sizes; (void)n_in; (void)out_size; (void)ws_size;
  const float* states = (const float*)d_in[0];
  const float* tptr   = (const float*)d_in[1];
  const float* dW1 = (const float*)d_in[2];
  const float* db1 = (const float*)d_in[3];
  const float* dW2 = (const float*)d_in[4];
  const float* db2 = (const float*)d_in[5];
  const float* dW3 = (const float*)d_in[6];
  const float* db3 = (const float*)d_in[7];
  const float* nW1 = (const float*)d_in[8];
  const float* nb1 = (const float*)d_in[9];
  const float* nW2 = (const float*)d_in[10];
  const float* nb2 = (const float*)d_in[11];
  const float* nW3 = (const float*)d_in[12];
  const float* nb3 = (const float*)d_in[13];
  float* out = (float*)d_out;

  size_t o = 0;
  char* base = (char*)d_ws;
  auto take = [&](size_t bytes) -> void* {
    void* r = base + o;
    o += (bytes + 255) & ~(size_t)255;
    return r;
  };
  f16* W1dh = (f16*)take(DIMX * HIDN * sizeof(f16));
  f16* W2dh = (f16*)take((size_t)HIDN * HIDN * sizeof(f16));
  f16* W3dh = (f16*)take(HIDN * DIMX * sizeof(f16));
  f16* W1nh = (f16*)take(DIMX * HIDN * sizeof(f16));
  f16* W2nh = (f16*)take((size_t)HIDN * HIDN * sizeof(f16));
  f16* W3nh = (f16*)take(HIDN * DIMX * sizeof(f16));
  f16* muH  = (f16*)take(BATCH * DIMX * sizeof(f16));
  f16* h1d  = (f16*)take((size_t)BATCH * HIDN * sizeof(f16));
  f16* t1d  = (f16*)take((size_t)BATCH * HIDN * sizeof(f16));
  f16* h2d  = (f16*)take((size_t)BATCH * HIDN * sizeof(f16));
  f16* t2d  = (f16*)take((size_t)BATCH * HIDN * sizeof(f16));
  f16* h1n  = (f16*)take((size_t)BATCH * HIDN * sizeof(f16));
  f16* h2n  = (f16*)take((size_t)BATCH * HIDN * sizeof(f16));
  float* b1d   = (float*)take(HIDN * sizeof(float));
  float* b1n   = (float*)take(HIDN * sizeof(float));
  float* Qdiag = (float*)take(BATCH * DIMX * sizeof(float));

  const dim3 blk(256);

  prep_kernel<<<dim3(2048), blk, 0, stream>>>(
      states, tptr, dW1, db1, dW2, dW3, nW1, nb1, nW2, nW3,
      muH, W1dh, W2dh, W3dh, W1nh, W2nh, W3nh, b1d, b1n);

  // Drift net
  wmma_gemm_kernel<<<dim3(HIDN / 128, BATCH / 128), blk, 0, stream>>>(
      muH, W1dh, b1d, h1d, t1d, nullptr, HIDN, DIMX, 0, 1);
  wmma_gemm_kernel<<<dim3(HIDN / 128, BATCH / 128), blk, 0, stream>>>(
      h1d, W2dh, db2, h2d, t2d, nullptr, HIDN, HIDN, 0, 1);
  wmma_gemm_kernel<<<dim3(DIMX / 128, BATCH / 128), blk, 0, stream>>>(
      h2d, W3dh, db3, nullptr, nullptr, out, DIMX, HIDN, 2 * DIMX, 0);  // mu_drift

  // Noise net
  wmma_gemm_kernel<<<dim3(HIDN / 128, BATCH / 128), blk, 0, stream>>>(
      muH, W1nh, b1n, h1n, nullptr, nullptr, HIDN, DIMX, 0, 1);
  wmma_gemm_kernel<<<dim3(HIDN / 128, BATCH / 128), blk, 0, stream>>>(
      h1n, W2nh, nb2, h2n, nullptr, nullptr, HIDN, HIDN, 0, 1);
  wmma_gemm_kernel<<<dim3(DIMX / 128, BATCH / 128), blk, 0, stream>>>(
      h2n, W3nh, nb3, nullptr, nullptr, Qdiag, DIMX, HIDN, DIMX, 2);    // Q_diag

  // Jacobian diagonal + sigma_drift + log_density_drift
  jac_kernel<<<dim3(BATCH), blk, 0, stream>>>(
      W1dh, W2dh, W3dh, t1d, t2d, Qdiag, states, out);
}
